// Net_33930241638743
// MI455X (gfx1250) — matrix operand order repaired and checked by reading, not compiled, not run on hardware
//
#include <hip/hip_runtime.h>
#include <math.h>

// ---------------------------------------------------------------------------
// MI455X (gfx1250). Compute-bound (~333 GFLOP vs ~150MB HBM traffic) ->
// v_wmma_f32_16x16x32_f16 everywhere (wave32, f32 accum).
// Register blocking M=32 x N=96 per wave (each LDS B fragment feeds 2 WMMAs),
// B fragments pinned into a 2-deep ds-read/WMMA software pipeline via
// sched_group_barrier, A fragments ping-ponged with a 2x-unrolled k-loop
// (no register copies). B tiles staged to LDS by the Tensor Data Mover:
// each wave TDMs its own 48-row quarter (per-wave s_wait_tensorcnt), double
// buffered, TDM hardware padding -> 40-half LDS rows (bank-conflict-free
// ds_load_b128 across 16 lanes).
// ---------------------------------------------------------------------------

#ifndef USE_TDM
#define USE_TDM 1
#endif

#if defined(__has_builtin)
#  if __has_builtin(__builtin_amdgcn_sched_group_barrier)
#    define SCHED_GB(m,n,id) __builtin_amdgcn_sched_group_barrier(m,n,id)
#  endif
#endif
#ifndef SCHED_GB
#  define SCHED_GB(m,n,id)
#endif

typedef _Float16 h16;
typedef unsigned int u32;
typedef __attribute__((ext_vector_type(16))) _Float16 v16h;
typedef __attribute__((ext_vector_type(8)))  float    v8f;
typedef __attribute__((ext_vector_type(4)))  u32      v4u;
typedef __attribute__((ext_vector_type(8)))  int      v8i;
typedef __attribute__((ext_vector_type(4)))  int      v4i;

union Frag16 { v16h v; uint4 u[2]; };

#define BB   8
#define CC   192
#define HH   32
#define WW   48
#define NPIX (BB*HH*WW)   // 12288
#define DIM  192
#define KM   384          // merged channels (192 y + 192 h)
#define BROW 40           // padded LDS row stride in halfs (32 data + 8 pad)

// ---------------------------------------------------------------------------
// TDM async 2-D tile load Global->LDS (D# per CDNA5 ISA).
// tile_dim0=32 halfs (64B rows), tile_dim1=rows, stride0=K halfs.
// pad_enable, pad_interval=16 dwords (code 3), pad_amount=4 dwords (code 3)
// => LDS row stride 40 halfs == BROW.
// ---------------------------------------------------------------------------
__device__ __forceinline__ void tdm_load_2d(u32 lds_off, const h16* gsrc,
                                            int tileRows, int strideHalfs)
{
    unsigned long long ga = (unsigned long long)(uintptr_t)gsrc;
    v4u g0 = { 1u,                                  // count=1 (valid user D#)
               lds_off,                             // lds_addr
               (u32)ga,                             // global_addr[31:0]
               ((u32)(ga >> 32) & 0x01FFFFFFu) | (2u << 30) };  // [56:32]|type=2
    v8i g1 = { (int)((1u << 16)                     // data_size = 2 bytes
                   | (1u << 20)                     // pad_enable
                   | (3u << 22)                     // pad_interval = 16 dwords
                   | (3u << 25)),                   // pad_amount   = 4 dwords
               0,                                   // atomic barrier addr = 0
               0x4000,                              // tensor_dim0 = 1<<30
               0x4000 | (32 << 16),                 // tensor_dim1 hi | tile_dim0
               tileRows,                            // tile_dim1
               strideHalfs,                         // tensor_dim0_stride
               0, 0 };
    v4i gz = { 0, 0, 0, 0 };
    asm volatile("tensor_load_to_lds %0, %1, %2, %3"
                 :: "s"(g0), "s"(g1), "s"(gz), "s"(gz) : "memory");
}

// scheduler pin: 12 ds-reads + 12 WMMAs as a 2-deep pipeline
__device__ __forceinline__ void pipe_pin()
{
    SCHED_GB(0x100, 4, 0);               // ds read x4 (two B frags in flight)
    #pragma unroll
    for (int i = 0; i < 4; ++i) {
        SCHED_GB(0x008, 2, 0);           // wmma x2
        SCHED_GB(0x100, 2, 0);           // ds read x2 (next B frag)
    }
    SCHED_GB(0x008, 4, 0);               // drain: wmma x4
}

// ---------------------------------------------------------------------------
// prep: quantize y (floor(x+0.5), clip [-128,127]) + h_tilde -> f16 NHWC
// ---------------------------------------------------------------------------
__global__ void prep_merged_kernel(const float* __restrict__ yin,
                                   const float* __restrict__ hin,
                                   h16* __restrict__ merged)
{
    int tid = blockIdx.x * blockDim.x + threadIdx.x;
    if (tid >= BB*KM*HH*WW) return;
    int x = tid % WW; int t = tid / WW;
    int y = t % HH;   t /= HH;
    int ch = t % KM;  int b = t / KM;
    float v;
    if (ch < CC) {
        v = yin[(((size_t)b*CC + ch)*HH + y)*WW + x];
        v = floorf(v + 0.5f);
        v = fminf(fmaxf(v, -128.0f), 127.0f);
    } else {
        v = hin[(((size_t)b*CC + (ch - CC))*HH + y)*WW + x];
    }
    merged[(((size_t)b*HH + y)*WW + x)*KM + ch] = (h16)v;
}

__global__ void prep_w_kernel(const float* __restrict__ src, h16* __restrict__ dst,
                              int OC, int IC)
{
    int tid = blockIdx.x * blockDim.x + threadIdx.x;
    if (tid >= OC*IC*9) return;
    int tap = tid % 9; int t = tid / 9;
    int ic = t % IC;   int o = t / IC;
    dst[((size_t)tap*OC + o)*IC + ic] = (h16)src[tid];
}

__global__ void prep_fc_kernel(const float* __restrict__ src, h16* __restrict__ dst)
{
    int tid = blockIdx.x * blockDim.x + threadIdx.x;
    if (tid >= 384*768) return;
    int k = tid % 768; int o = tid / 768;
    int ch = k >> 2;   int pos = k & 3;
    dst[(size_t)o*768 + pos*DIM + ch] = (h16)src[tid];
}

// ---------------------------------------------------------------------------
// conv0: block = 4 pixels; wave = 2 adjacent pixels (M=32) x 96 cols (N).
// Rows = 16 output positions per pixel, K = 9 taps x 384 merged channels.
// block_sample's masked padded gather folded into A addresses.
// ---------------------------------------------------------------------------
__global__ __launch_bounds__(128) void conv0_kernel(
    const h16* __restrict__ merged, const h16* __restrict__ w0r,
    const float* __restrict__ b0, h16* __restrict__ t0)
{
    __shared__ __align__(16) h16 Blds[2][DIM*BROW];
    const int tid  = threadIdx.x;
    const int wave = tid >> 5;
    const int lane = tid & 31;
    const int l16  = lane & 15;
    const bool hi  = lane >= 16;
    const int koff8  = hi ? 16 : 0;   // A: 16 consecutive K per lane
    const int koff16 = hi ? 16 : 0;   // B: 16 consecutive K per lane

    const int pr    = wave >> 1;           // patch pair within block
    const int colh  = wave & 1;            // column half (0 or 1) -> 96 cols
    const int nA    = blockIdx.x*4 + pr*2;
    const int nB    = nA + 1;
    const int bA = nA/(HH*WW), rA = nA%(HH*WW), yA = rA/WW, xA = rA%WW;
    const int bX = nB/(HH*WW), rX = nB%(HH*WW), yB = rX/WW, xB = rX%WW;
    const int oy = l16 >> 2, ox = l16 & 3;

    v8f acc0[6] = {};
    v8f acc1[6] = {};

    const u32 lds_base = (u32)(uintptr_t)(&Blds[0][0]);
    const u32 lds_sz   = (u32)(DIM*BROW*2);

    // each wave stages its own 48-row quarter of the B tile
    auto stage = [&](int s, int buf) {
        const int tap = s / 12, kc = s % 12;
#if USE_TDM
        const h16* src = w0r + ((size_t)(tap*DIM + wave*48))*KM + kc*32;
        tdm_load_2d(lds_base + (u32)buf*lds_sz + (u32)(wave*48*BROW*2),
                    src, 48, KM);
#else
        const h16* src = w0r + ((size_t)tap*DIM)*KM + kc*32;
        #pragma unroll
        for (int i = 0; i < 6; ++i) {
            int idx = tid + i*128;
            int c = idx >> 2, chunk = idx & 3;
            ((uint4*)(Blds[buf] + c*BROW))[chunk] =
                ((const uint4*)(src + (size_t)c*KM))[chunk];
        }
#endif
    };
    auto stage_wait = [&]() {
#if USE_TDM
        __builtin_amdgcn_s_wait_tensorcnt(0);   // wait own quarter
#endif
        __syncthreads();
    };
    auto loadA = [&](Frag16& a, int s, int bi, int y0, int x0) {
        const int tap = s / 12, kc = s % 12;
        const int di = tap / 3, dj = tap % 3;
        const int pi = oy + di - 1, pj = ox + dj - 1;   // patch coords
        const int yy = y0 + pi - 3, xx = x0 + pj - 2;   // image coords
        bool valid = ((unsigned)pi < 4u) & ((unsigned)pj < 4u)
                   & ((unsigned)yy < (unsigned)HH) & ((unsigned)xx < (unsigned)WW);
        if (pi == 3 && pj >= 2 && kc < 6) valid = false; // AR mask (y channels)
        if (valid) {
            const h16* ap = merged + ((size_t)((bi*HH + yy)*WW + xx))*KM
                                   + kc*32 + koff8;
            a.u[0] = *(const uint4*)(ap);
            a.u[1] = *(const uint4*)(ap + 8);
        } else {
            a.u[0] = make_uint4(0u,0u,0u,0u);
            a.u[1] = make_uint4(0u,0u,0u,0u);
        }
    };
    auto mm = [&](const Frag16& fA, const Frag16& fB, int buf) {
        const h16* bb = Blds[buf] + (colh*96 + l16)*BROW + koff16;
        Frag16 bf[2];
        bf[0].u[0] = *(const uint4*)(bb);
        bf[0].u[1] = *(const uint4*)(bb + 8);
        #pragma unroll
        for (int ct = 0; ct < 6; ++ct) {
            if (ct + 1 < 6) {
                const h16* bp = bb + (ct+1)*16*BROW;
                bf[(ct+1)&1].u[0] = *(const uint4*)(bp);
                bf[(ct+1)&1].u[1] = *(const uint4*)(bp + 8);
            }
            acc0[ct] = __builtin_amdgcn_wmma_f32_16x16x32_f16(
                false, fA.v, false, bf[ct&1].v, (short)0, acc0[ct], false, false);
            acc1[ct] = __builtin_amdgcn_wmma_f32_16x16x32_f16(
                false, fB.v, false, bf[ct&1].v, (short)0, acc1[ct], false, false);
        }
        pipe_pin();   // enforce 2-deep ds-read/wmma pipeline in this region
    };

    Frag16 a0A, a0B, a1A, a1B;          // ping-pong A fragments (no copies)
    loadA(a0A, 0, bA, yA, xA);
    loadA(a0B, 0, bX, yB, xB);
    stage(0, 0);
    stage_wait();

    int s = 0;
    for (; s + 2 < 108; s += 2) {
        stage(s + 1, 1);
        loadA(a1A, s + 1, bA, yA, xA);
        loadA(a1B, s + 1, bX, yB, xB);
        mm(a0A, a0B, 0);
        stage_wait();
        stage(s + 2, 0);
        loadA(a0A, s + 2, bA, yA, xA);
        loadA(a0B, s + 2, bX, yB, xB);
        mm(a1A, a1B, 1);
        stage_wait();
    }
    // peeled last pair (s == 106)
    stage(107, 1);
    loadA(a1A, 107, bA, yA, xA);
    loadA(a1B, 107, bX, yB, xB);
    mm(a0A, a0B, 0);
    stage_wait();
    mm(a1A, a1B, 1);

    // epilogue: +bias, leaky ReLU, store t0[n][pos][oc] f16
    #pragma unroll
    for (int ct = 0; ct < 6; ++ct) {
        const int o = colh*96 + ct*16 + l16;
        const float bo = b0[o];
        #pragma unroll
        for (int v = 0; v < 8; ++v) {
            const int pout = v + (hi ? 8 : 0);
            float v0 = acc0[ct][v] + bo;
            v0 = v0 >= 0.0f ? v0 : 0.2f*v0;
            t0[((size_t)nA*16 + pout)*DIM + o] = (h16)v0;
            float v1 = acc1[ct][v] + bo;
            v1 = v1 >= 0.0f ? v1 : 0.2f*v1;
            t0[((size_t)nB*16 + pout)*DIM + o] = (h16)v1;
        }
    }
}

// ---------------------------------------------------------------------------
// conv1 (INS=4, stride2) / conv2 (INS=2, stride1): block = 16 patches;
// wave = 2 groups of 4 patches (M=32 rows) x 96 cols. Output [n][4][192].
// ---------------------------------------------------------------------------
template<int INS, int STRIDE>
__global__ __launch_bounds__(128) void conv_s_kernel(
    const h16* __restrict__ tin, const h16* __restrict__ wr,
    const float* __restrict__ bias, h16* __restrict__ tout)
{
    __shared__ __align__(16) h16 Blds[2][DIM*BROW];
    const int tid  = threadIdx.x;
    const int wave = tid >> 5;
    const int lane = tid & 31;
    const int l16  = lane & 15;
    const bool hi  = lane >= 16;
    const int koff8  = hi ? 16 : 0;
    const int koff16 = hi ? 16 : 0;

    const int pr   = wave >> 1;
    const int colh = wave & 1;
    const int gA = blockIdx.x*4 + pr*2;   // two groups of 4 patches
    const int gB = gA + 1;
    const int q = l16 >> 2;               // patch within group
    const int opos = l16 & 3;
    const int oy = opos >> 1, ox = opos & 1;

    v8f acc0[6] = {};
    v8f acc1[6] = {};
    constexpr int KC = DIM/32;   // 6
    constexpr int NS = 9*KC;     // 54

    const u32 lds_base = (u32)(uintptr_t)(&Blds[0][0]);
    const u32 lds_sz   = (u32)(DIM*BROW*2);

    auto stage = [&](int s, int buf) {
        const int tap = s / KC, kc = s % KC;
#if USE_TDM
        const h16* src = wr + ((size_t)(tap*DIM + wave*48))*DIM + kc*32;
        tdm_load_2d(lds_base + (u32)buf*lds_sz + (u32)(wave*48*BROW*2),
                    src, 48, DIM);
#else
        const h16* src = wr + ((size_t)tap*DIM)*DIM + kc*32;
        #pragma unroll
        for (int i = 0; i < 6; ++i) {
            int idx = tid + i*128;
            int c = idx >> 2, chunk = idx & 3;
            ((uint4*)(Blds[buf] + c*BROW))[chunk] =
                ((const uint4*)(src + (size_t)c*DIM))[chunk];
        }
#endif
    };
    auto stage_wait = [&]() {
#if USE_TDM
        __builtin_amdgcn_s_wait_tensorcnt(0);
#endif
        __syncthreads();
    };
    auto loadA = [&](Frag16& a, int s, int g) {
        const int tap = s / KC, kc = s % KC;
        const int di = tap/3, dj = tap%3;
        const int iy = oy*STRIDE + di - 1, ix = ox*STRIDE + dj - 1;
        const bool valid = ((unsigned)iy < (unsigned)INS)
                         & ((unsigned)ix < (unsigned)INS);
        if (valid) {
            const int n = g*4 + q;
            const h16* ap = tin + ((size_t)n*(INS*INS) + iy*INS + ix)*DIM
                                + kc*32 + koff8;
            a.u[0] = *(const uint4*)(ap);
            a.u[1] = *(const uint4*)(ap + 8);
        } else {
            a.u[0] = make_uint4(0u,0u,0u,0u);
            a.u[1] = make_uint4(0u,0u,0u,0u);
        }
    };
    auto mm = [&](const Frag16& fA, const Frag16& fB, int buf) {
        const h16* bb = Blds[buf] + (colh*96 + l16)*BROW + koff16;
        Frag16 bf[2];
        bf[0].u[0] = *(const uint4*)(bb);
        bf[0].u[1] = *(const uint4*)(bb + 8);
        #pragma unroll
        for (int ct = 0; ct < 6; ++ct) {
            if (ct + 1 < 6) {
                const h16* bp = bb + (ct+1)*16*BROW;
                bf[(ct+1)&1].u[0] = *(const uint4*)(bp);
                bf[(ct+1)&1].u[1] = *(const uint4*)(bp + 8);
            }
            acc0[ct] = __builtin_amdgcn_wmma_f32_16x16x32_f16(
                false, fA.v, false, bf[ct&1].v, (short)0, acc0[ct], false, false);
            acc1[ct] = __builtin_amdgcn_wmma_f32_16x16x32_f16(
                false, fB.v, false, bf[ct&1].v, (short)0, acc1[ct], false, false);
        }
        pipe_pin();
    };

    Frag16 a0A, a0B, a1A, a1B;
    loadA(a0A, 0, gA);
    loadA(a0B, 0, gB);
    stage(0, 0);
    stage_wait();

    int s = 0;
    for (; s + 2 < NS; s += 2) {
        stage(s + 1, 1);
        loadA(a1A, s + 1, gA);
        loadA(a1B, s + 1, gB);
        mm(a0A, a0B, 0);
        stage_wait();
        stage(s + 2, 0);
        loadA(a0A, s + 2, gA);
        loadA(a0B, s + 2, gB);
        mm(a1A, a1B, 1);
        stage_wait();
    }
    stage(NS - 1, 1);
    loadA(a1A, NS - 1, gA);
    loadA(a1B, NS - 1, gB);
    mm(a0A, a0B, 0);
    stage_wait();
    mm(a1A, a1B, 1);

    #pragma unroll
    for (int ct = 0; ct < 6; ++ct) {
        const int o = colh*96 + ct*16 + l16;
        const float bo = bias[o];
        #pragma unroll
        for (int v = 0; v < 8; ++v) {
            const int pout = v + (hi ? 8 : 0);
            const int qo = pout >> 2, op = pout & 3;
            float v0 = acc0[ct][v] + bo;
            v0 = v0 >= 0.0f ? v0 : 0.2f*v0;
            tout[((size_t)(gA*4 + qo)*4 + op)*DIM + o] = (h16)v0;
            float v1 = acc1[ct][v] + bo;
            v1 = v1 >= 0.0f ? v1 : 0.2f*v1;
            tout[((size_t)(gB*4 + qo)*4 + op)*DIM + o] = (h16)v1;
        }
    }
}

// ---------------------------------------------------------------------------
// fc: ctx[n][384] = t2[n][768] @ fc16^T + fc_b.
// Block = 32 rows; wave = 32 rows (2 A frags) x 96-col quarter.
// ---------------------------------------------------------------------------
__global__ __launch_bounds__(128) void fc_kernel(
    const h16* __restrict__ t2, const h16* __restrict__ fcw,
    const float* __restrict__ fcb, float* __restrict__ ctx)
{
    const int tid  = threadIdx.x;
    const int wave = tid >> 5;              // column quarter
    const int lane = tid & 31;
    const int l16  = lane & 15;
    const bool hi  = lane >= 16;
    const int koff8  = hi ? 16 : 0;
    const int koff16 = hi ? 16 : 0;
    const int m0 = blockIdx.x*2;            // two 16-row tiles

    v8f acc0[6] = {};
    v8f acc1[6] = {};
    const h16* arow0 = t2 + (size_t)(m0*16 + l16)*768;
    const h16* arow1 = arow0 + (size_t)16*768;
    const h16* bcol  = fcw + (size_t)(wave*96 + l16)*768 + koff16;

    for (int kk = 0; kk < 24; ++kk) {
        Frag16 a0, a1;
        const h16* ap0 = arow0 + kk*32 + koff8;
        a0.u[0] = *(const uint4*)(ap0);
        a0.u[1] = *(const uint4*)(ap0 + 8);
        const h16* ap1 = arow1 + kk*32 + koff8;
        a1.u[0] = *(const uint4*)(ap1);
        a1.u[1] = *(const uint4*)(ap1 + 8);
        Frag16 bf[2];
        {
            const h16* bp = bcol + kk*32;
            bf[0].u[0] = *(const uint4*)(bp);
            bf[0].u[1] = *(const uint4*)(bp + 8);
        }
        #pragma unroll
        for (int ct = 0; ct < 6; ++ct) {
            if (ct + 1 < 6) {
                const h16* bp = bcol + (size_t)(ct+1)*16*768 + kk*32;
                bf[(ct+1)&1].u[0] = *(const uint4*)(bp);
                bf[(ct+1)&1].u[1] = *(const uint4*)(bp + 8);
            }
            acc0[ct] = __builtin_amdgcn_wmma_f32_16x16x32_f16(
                false, a0.v, false, bf[ct&1].v, (short)0, acc0[ct], false, false);
            acc1[ct] = __builtin_amdgcn_wmma_f32_16x16x32_f16(
                false, a1.v, false, bf[ct&1].v, (short)0, acc1[ct], false, false);
        }
    }
    #pragma unroll
    for (int ct = 0; ct < 6; ++ct) {
        const int o = wave*96 + ct*16 + l16;
        const float bo = fcb[o];
        #pragma unroll
        for (int v = 0; v < 8; ++v) {
            const int row0 = m0*16 + v + (hi ? 8 : 0);
            ctx[(size_t)row0*384 + o] = acc0[ct][v] + bo;
            ctx[(size_t)(row0 + 16)*384 + o] = acc1[ct][v] + bo;
        }
    }
}

// ---------------------------------------------------------------------------
// Gaussian likelihood: clip(ndtr(u) - ndtr(l), 1e-12), output NCHW f32
// ---------------------------------------------------------------------------
__global__ void likelihood_kernel(const h16* __restrict__ merged,
                                  const float* __restrict__ ctx,
                                  float* __restrict__ out)
{
    int tid = blockIdx.x * blockDim.x + threadIdx.x;
    if (tid >= BB*CC*HH*WW) return;
    int x = tid % WW; int t = tid / WW;
    int y = t % HH;   t /= HH;
    int c = t % CC;   int b = t / CC;
    int np = (b*HH + y)*WW + x;
    float yq = (float)merged[(size_t)np*KM + c];
    float mu = ctx[(size_t)np*384 + c];
    float sg = expf(ctx[(size_t)np*384 + 192 + c]);
    float inv = 1.0f / sg;
    const float RS2 = 0.70710678118654752f;  // 1/sqrt(2)
    float u = (yq - mu + 0.5f) * inv;
    float l = (yq - mu - 0.5f) * inv;
    float lik = 0.5f * (erfcf(-u*RS2) - erfcf(-l*RS2));
    out[tid] = fmaxf(lik, 1e-12f);
}

// ---------------------------------------------------------------------------
extern "C" void kernel_launch(void* const* d_in, const int* in_sizes, int n_in,
                              void* d_out, int out_size, void* d_ws, size_t ws_size,
                              hipStream_t stream)
{
    const float* y   = (const float*)d_in[0];
    const float* ht  = (const float*)d_in[1];
    const float* w0  = (const float*)d_in[2];
    const float* b0  = (const float*)d_in[3];
    const float* w1  = (const float*)d_in[4];
    const float* b1  = (const float*)d_in[5];
    const float* w2  = (const float*)d_in[6];
    const float* b2  = (const float*)d_in[7];
    const float* fcw = (const float*)d_in[8];
    const float* fcb = (const float*)d_in[9];
    float* out = (float*)d_out;

    char* p = (char*)d_ws;
    auto take = [&](size_t bytes) {
        char* r = p; p += (bytes + 255) & ~((size_t)255); return r;
    };
    h16* merged = (h16*)take((size_t)NPIX*KM*2);       //  9.4 MB
    h16* w0r    = (h16*)take((size_t)9*DIM*KM*2);      //  1.3 MB
    h16* w1r    = (h16*)take((size_t)9*DIM*DIM*2);     //  0.7 MB
    h16* w2r    = (h16*)take((size_t)9*DIM*DIM*2);     //  0.7 MB
    h16* fc16   = (h16*)take((size_t)384*768*2);       //  0.6 MB
    h16* t0     = (h16*)take((size_t)NPIX*16*DIM*2);   // 75.5 MB
    h16* t1     = (h16*)take((size_t)NPIX*4*DIM*2);    // 18.9 MB
    h16* t2     = (h16*)take((size_t)NPIX*4*DIM*2);    // 18.9 MB
    float* ctx  = (float*)take((size_t)NPIX*384*4);    // 18.9 MB

    prep_merged_kernel<<<(BB*KM*HH*WW + 255)/256, 256, 0, stream>>>(y, ht, merged);
    prep_w_kernel<<<(DIM*KM*9 + 255)/256, 256, 0, stream>>>(w0, w0r, DIM, KM);
    prep_w_kernel<<<(DIM*DIM*9 + 255)/256, 256, 0, stream>>>(w1, w1r, DIM, DIM);
    prep_w_kernel<<<(DIM*DIM*9 + 255)/256, 256, 0, stream>>>(w2, w2r, DIM, DIM);
    prep_fc_kernel<<<(384*768 + 255)/256, 256, 0, stream>>>(fcw, fc16);

    conv0_kernel<<<NPIX/4, 128, 0, stream>>>(merged, w0r, b0, t0);
    conv_s_kernel<4,2><<<NPIX/16, 128, 0, stream>>>(t0, w1r, b1, t1);
    conv_s_kernel<2,1><<<NPIX/16, 128, 0, stream>>>(t1, w2r, b2, t2);
    fc_kernel<<<NPIX/32, 128, 0, stream>>>(t2, fc16, fcb, ctx);

    likelihood_kernel<<<(BB*CC*HH*WW + 255)/256, 256, 0, stream>>>(merged, ctx, out);
}
